// SE3TransformerInteractionBlock_51058571215445
// MI455X (gfx1250) — matrix operand order repaired and checked by reading, not compiled, not run on hardware
//
#include <hip/hip_runtime.h>
#include <hip/hip_bf16.h>

#define NN 20000
#define EE 100000
#define CC 32
#define HHD 4
#define DHH 8
#define NBB 16
#define HIDD 64
#define WNN 1024

typedef __attribute__((ext_vector_type(16))) _Float16 v16h;
typedef __attribute__((ext_vector_type(2)))  _Float16 v2h;
typedef __attribute__((ext_vector_type(8)))  float    v8f;
typedef __attribute__((ext_vector_type(4)))  float    f4;

// fast sigmoid: v_exp + v_add + v_rcp (approximate rcp, ~1 ulp)
__device__ __forceinline__ float sigm(float x) {
  return __builtin_amdgcn_rcpf(1.0f + __expf(-x));
}

// monotone float -> uint map so atomicMax(u32) == float max (init value 0 == -inf)
__device__ __forceinline__ unsigned orderKey(float f) {
  unsigned u = __float_as_uint(f);
  return (u & 0x80000000u) ? ~u : (u | 0x80000000u);
}
__device__ __forceinline__ float decodeKey(unsigned k) {
  return (k & 0x80000000u) ? __uint_as_float(k ^ 0x80000000u) : __uint_as_float(~k);
}

// ---------------- Kernel A: init scratch + q = (X @ w_q)/sqrt(C) -----------
__global__ void k_node_q_init(const float* __restrict__ nf, const float* __restrict__ wq,
                              float* __restrict__ q, unsigned* __restrict__ mArr,
                              float* __restrict__ den, float* __restrict__ agg) {
  int n = blockIdx.x * (blockDim.x >> 5) + (threadIdx.x >> 5);
  int k = threadIdx.x & 31;
  if (n >= NN) return;
  float x = nf[n * CC + k];
  float acc = 0.f;
#pragma unroll
  for (int i = 0; i < CC; ++i) acc = __builtin_fmaf(__shfl(x, i), wq[i * CC + k], acc);
  q[n * CC + k]   = acc * 0.17677669529663687f;   // 1/sqrt(32)
  agg[n * CC + k] = 0.f;
  if (k < HHD) { mArr[n * HHD + k] = 0u; den[n * HHD + k] = 0.f; }
}

// ---------------- Kernel B: fused radial-MLP + TP contraction (WMMA) -------
// k/v_edge[e,:] = (hid[e,:] (+bias row)) @ Bmat, Bmat[(h,i),k] = W2[h, i*32+k]
// A row = outer(hid, x_src): per K=32 step s, A-frag = xfrag * hid[m][s].
// Hidden layer itself is a 16x64 (K=16 zero-padded to 32) WMMA with b1 in C.
template <int KPATH>
__global__ __launch_bounds__(512) void k_edge_gemm(
    const float* __restrict__ nf, const int* __restrict__ ei,
    const float* __restrict__ shp, const float* __restrict__ emb,
    const float* __restrict__ elen, const void* __restrict__ mrp,
    const float* __restrict__ w1, const float* __restrict__ b1,
    const float* __restrict__ w2, const float* __restrict__ b2,
    const float* __restrict__ wdot, const float* __restrict__ q,
    float* __restrict__ logits, unsigned* __restrict__ mArr,
    float* __restrict__ vws) {
  constexpr int WAVES = 16;
  __shared__ _Float16 sB[(HIDD + 1) * 2 * 32 * 16];  // 66560 halves = 130 KB, W2 B-frags
  __shared__ _Float16 sW1F[4 * 32 * 16];             // W1 B-frags (K=32, upper 16 zero)
  __shared__ float sB1[HIDD];
  __shared__ _Float16 sHidH[WAVES][16 * 66];         // per-wave hidden, f16
  __shared__ float sWd[KPATH ? 64 : 1];
  __shared__ float sKt[KPATH ? WAVES * 16 * 33 : 1];

  const int tid = threadIdx.x;
  // W2 (+bias slab) fragments in WMMA B-operand layout (f16).
  for (int j = tid; j < (HIDD + 1) * WNN; j += 512) {
    int s = j >> 10, rem = j & 1023;
    int i = rem >> 5, nn = rem & 31;
    int tnt = nn >> 4, Nl = nn & 15;
    int bl = Nl + ((i & 8) << 1);      // lanes 0-15: K {0-7,16-23}, lanes 16-31: K {8-15,24-31}
    int hf = (i & 7) | ((i >> 1) & 8);
    float v = (s < HIDD) ? w2[j] : b2[rem];
    sB[(((s * 2 + tnt) * 32 + bl) << 4) + hf] = (_Float16)v;
  }
  // W1 fragments: 4 N-tiles, single K=32 chunk; K-rows 16..31 are zero.
  for (int j = tid; j < 4 * 32 * 16; j += 512) {
    int t1 = j >> 9;
    int bl = (j >> 4) & 31;
    int hf = j & 15;
    int i = (hf & 7) + ((bl & 16) >> 1) + ((hf & 8) << 1);
    int nj = t1 * 16 + (bl & 15);
    sW1F[j] = (i < NBB) ? (_Float16)w1[i * HIDD + nj] : (_Float16)0.f;
  }
  if (tid < HIDD) sB1[tid] = b1[tid];
  if (KPATH && tid < 64) sWd[tid] = wdot[tid];
  __syncthreads();

  float inv_mr = 1.f;
  if (KPATH) {  // max_radius: tolerate int or float encoding of the scalar
    int bits = *(const int*)mrp;
    float mr = (bits > 0 && bits < 0x00800000) ? (float)bits : __int_as_float(bits);
    inv_mr = 1.f / mr;
  }

  const int wid = tid >> 5, lane = tid & 31;
  const int nTiles = EE / 16;
  for (int tile = blockIdx.x * WAVES + wid; tile < nTiles; tile += gridDim.x * WAVES) {
    const int e0 = tile * 16;
    const int m = lane & 15;
    const int e = e0 + m;
    const int srcn = ei[e];
    const float shv = shp[e];

    // x_src in WMMA A-fragment layout: lane<16: K{0-7,16-23}; lane>=16: K{8-15,24-31}
    const int i0 = (lane < 16) ? 0 : 8;
    const float* xr = nf + (size_t)srcn * CC;
    f4 xa = *(const f4*)(xr + i0);
    f4 xb = *(const f4*)(xr + i0 + 4);
    f4 xc = *(const f4*)(xr + i0 + 16);
    f4 xd = *(const f4*)(xr + i0 + 20);
    v16h xf;
#pragma unroll
    for (int t = 0; t < 4; ++t) {
      xf[t]      = (_Float16)(xa[t] * shv);
      xf[4 + t]  = (_Float16)(xb[t] * shv);
      xf[8 + t]  = (_Float16)(xc[t] * shv);
      xf[12 + t] = (_Float16)(xd[t] * shv);
    }

    // hidden = silu(emb @ W1 + b1) via WMMA: A = emb (K=16 real + 16 zero).
    // Issue all 4 N-tile WMMAs first (independent accumulators), then silu.
    const float* er = emb + (size_t)e * NBB;
    f4 ea = *(const f4*)(er + i0);
    f4 eb = *(const f4*)(er + i0 + 4);
    v16h aemb = {};
#pragma unroll
    for (int t = 0; t < 4; ++t) {
      aemb[t]     = (_Float16)ea[t];
      aemb[4 + t] = (_Float16)eb[t];
    }
    _Float16* hbase = &sHidH[wid][0];
    const int jcol = lane & 15;
    const int mhi = (lane >> 4) << 3;
    v8f ha[4];
#pragma unroll
    for (int t1 = 0; t1 < 4; ++t1) {
      float bj = sB1[t1 * 16 + jcol];
      v8f c0 = {bj, bj, bj, bj, bj, bj, bj, bj};
      const v16h bw = *(const v16h*)&sW1F[(t1 * 32 + lane) << 4];
      ha[t1] = __builtin_amdgcn_wmma_f32_16x16x32_f16(false, aemb, false, bw, (short)0, c0, false, false);
    }
#pragma unroll
    for (int t1 = 0; t1 < 4; ++t1) {
      int j = t1 * 16 + jcol;
#pragma unroll
      for (int r = 0; r < 8; ++r) {
        float hv = ha[t1][r];
        hbase[(r + mhi) * 66 + j] = (_Float16)(hv * sigm(hv));
      }
    }

    // (16 x 2080) @ (2080 x 32) GEMM; 2 K-steps per iteration, bias epilogue.
    v8f acc0 = {}; v8f acc1 = {};
    const _Float16* hrow = &sHidH[wid][m * 66];
    for (int s2 = 0; s2 < HIDD / 2; ++s2) {
      v2h hp = *(const v2h*)&hrow[2 * s2];
      _Float16 h0 = hp[0], h1 = hp[1];
      v16h af0, af1;
#pragma unroll
      for (int t = 0; t < 16; ++t) { af0[t] = xf[t] * h0; af1[t] = xf[t] * h1; }
      const v16h b00 = *(const v16h*)&sB[(((2 * s2) * 2)     * 32 + lane) << 4];
      const v16h b01 = *(const v16h*)&sB[(((2 * s2) * 2 + 1) * 32 + lane) << 4];
      const v16h b10 = *(const v16h*)&sB[(((2 * s2 + 1) * 2)     * 32 + lane) << 4];
      const v16h b11 = *(const v16h*)&sB[(((2 * s2 + 1) * 2 + 1) * 32 + lane) << 4];
      acc0 = __builtin_amdgcn_wmma_f32_16x16x32_f16(false, af0, false, b00, (short)0, acc0, false, false);
      acc1 = __builtin_amdgcn_wmma_f32_16x16x32_f16(false, af0, false, b01, (short)0, acc1, false, false);
      acc0 = __builtin_amdgcn_wmma_f32_16x16x32_f16(false, af1, false, b10, (short)0, acc0, false, false);
      acc1 = __builtin_amdgcn_wmma_f32_16x16x32_f16(false, af1, false, b11, (short)0, acc1, false, false);
    }
    {  // bias row of the GEMM: hid == 1.0 -> A-frag is xf itself
      const v16h bb0 = *(const v16h*)&sB[((HIDD * 2)     * 32 + lane) << 4];
      const v16h bb1 = *(const v16h*)&sB[((HIDD * 2 + 1) * 32 + lane) << 4];
      acc0 = __builtin_amdgcn_wmma_f32_16x16x32_f16(false, xf, false, bb0, (short)0, acc0, false, false);
      acc1 = __builtin_amdgcn_wmma_f32_16x16x32_f16(false, xf, false, bb1, (short)0, acc1, false, false);
    }
    const float tpn = 0.17677669529663687f;  // 1/sqrt(C*1)

    if constexpr (!KPATH) {
      // store v_edge tile from C/D layout: VGPR r, lane -> (row, col)
#pragma unroll
      for (int r = 0; r < 8; ++r) {
        int row = r + mhi;
        float* vp = vws + (size_t)(e0 + row) * CC;
        vp[jcol]      = acc0[r] * tpn;
        vp[jcol + 16] = acc1[r] * tpn;
      }
    } else {
      float* kt = &sKt[wid * (16 * 33)];
#pragma unroll
      for (int r = 0; r < 8; ++r) {
        int row = r + mhi;
        kt[row * 33 + jcol]      = acc0[r] * tpn;
        kt[row * 33 + jcol + 16] = acc1[r] * tpn;
      }
      // logits: 64 (edge,head) pairs per tile, 2 per lane
#pragma unroll
      for (int pp = 0; pp < 2; ++pp) {
        int p = lane + pp * 32;
        int me = p >> 2, h = p & 3;
        int ee2 = e0 + me;
        int dstn = ei[EE + ee2];
        const float* qp = q + (size_t)dstn * CC + h * DHH;
        const float* kp = &kt[me * 33 + h * DHH];
        float lg = 0.f;
#pragma unroll
        for (int i = 0; i < 8; ++i) {
          float t2 = 0.f;
#pragma unroll
          for (int j2 = 0; j2 < 8; ++j2) t2 = __builtin_fmaf(sWd[i * 8 + j2], kp[j2], t2);
          lg = __builtin_fmaf(qp[i], t2, lg);
        }
        float cut = sigm(10.f * (1.f - elen[ee2] * inv_mr));
        lg *= 0.04419417382415922f * cut;  // (1/DH) * (1/sqrt(DH)) * cutoff
        logits[(size_t)ee2 * HHD + h] = lg;
        atomicMax(&mArr[(size_t)dstn * HHD + h], orderKey(lg));
      }
    }
  }
}

// ---------------- Kernel C: exp/segment-sum accumulation -------------------
__global__ void k_edge_pass2(const int* __restrict__ ei, const float* __restrict__ logits,
                             const unsigned* __restrict__ mArr, const float* __restrict__ vws,
                             float* __restrict__ den, float* __restrict__ agg) {
  int t = blockIdx.x * blockDim.x + threadIdx.x;
  if (t >= EE * HHD) return;
  int e = t >> 2, h = t & 3;
  int dstn = ei[EE + e];
  float mv = decodeKey(mArr[dstn * HHD + h]);
  float ex = __expf(logits[(size_t)e * HHD + h] - mv);
  atomicAdd(&den[dstn * HHD + h], ex);
  const float* vp = vws + (size_t)e * CC + h * DHH;
  float* ap = agg + (size_t)dstn * CC + h * DHH;
#pragma unroll
  for (int j = 0; j < DHH; ++j) atomicAdd(&ap[j], ex * vp[j]);
}

// ---------------- Kernel D: node epilogue (out proj + skip + FFN) ----------
__global__ void k_node_out(const float* __restrict__ nf, const float* __restrict__ den,
                           const float* __restrict__ agg, const float* __restrict__ wout,
                           const float* __restrict__ fw1, const float* __restrict__ fw2,
                           float* __restrict__ out) {
  int n = blockIdx.x * (blockDim.x >> 5) + (threadIdx.x >> 5);
  int k = threadIdx.x & 31;
  if (n >= NN) return;
  float dh = den[n * HHD + (k >> 3)];
  float a = (dh > 0.f) ? agg[n * CC + k] * __builtin_amdgcn_rcpf(dh) : 0.f;
  float o = 0.f;
#pragma unroll
  for (int i = 0; i < CC; ++i) o = __builtin_fmaf(__shfl(a, i), wout[i * CC + k], o);
  float ao = nf[n * CC + k] + o * 0.17677669529663687f;
  float h1 = 0.f, h2 = 0.f;
#pragma unroll
  for (int i = 0; i < CC; ++i) {
    float v = __shfl(ao, i);
    h1 = __builtin_fmaf(v, fw1[i * 64 + k], h1);
    h2 = __builtin_fmaf(v, fw1[i * 64 + k + 32], h2);
  }
  h1 *= 0.17677669529663687f;
  h2 *= 0.17677669529663687f;
  float g1 = h1 * sigm(fabsf(h1));
  float g2 = h2 * sigm(fabsf(h2));
  float f = 0.f;
#pragma unroll
  for (int i = 0; i < CC; ++i) {
    f = __builtin_fmaf(__shfl(g1, i), fw2[i * CC + k], f);
    f = __builtin_fmaf(__shfl(g2, i), fw2[(i + 32) * CC + k], f);
  }
  out[n * CC + k] = ao + f * 0.125f;  // 1/sqrt(2C)
}

extern "C" void kernel_launch(void* const* d_in, const int* in_sizes, int n_in,
                              void* d_out, int out_size, void* d_ws, size_t ws_size,
                              hipStream_t stream) {
  (void)in_sizes; (void)n_in; (void)out_size; (void)ws_size;
  const float* nf   = (const float*)d_in[0];
  const int*   ei   = (const int*)d_in[1];
  const float* shp  = (const float*)d_in[2];
  const float* emb  = (const float*)d_in[3];
  const float* elen = (const float*)d_in[4];
  const void*  mrp  = d_in[5];
  const float* wq   = (const float*)d_in[6];
  const float* kw1  = (const float*)d_in[7];
  const float* kb1  = (const float*)d_in[8];
  const float* kw2  = (const float*)d_in[9];
  const float* kb2  = (const float*)d_in[10];
  const float* vw1  = (const float*)d_in[11];
  const float* vb1  = (const float*)d_in[12];
  const float* vw2  = (const float*)d_in[13];
  const float* vb2  = (const float*)d_in[14];
  const float* wdot = (const float*)d_in[15];
  const float* wout = (const float*)d_in[16];
  const float* fw1  = (const float*)d_in[17];
  const float* fw2  = (const float*)d_in[18];

  float* ws = (float*)d_ws;
  float*    q      = ws;                             // N*C
  unsigned* mArr   = (unsigned*)(ws + NN * CC);      // N*H (uint keys)
  float*    den    = ws + NN * CC + NN * HHD;        // N*H
  float*    agg    = den + NN * HHD;                 // N*C
  float*    logits = agg + NN * CC;                  // E*H
  float*    vws    = logits + (size_t)EE * HHD;      // E*C   (total ~20.2 MB)

  k_node_q_init<<<(NN + 7) / 8, 256, 0, stream>>>(nf, wq, q, mArr, den, agg);
  k_edge_gemm<0><<<200, 512, 0, stream>>>(nf, ei, shp, emb, elen, mrp,
                                          vw1, vb1, vw2, vb2, nullptr, nullptr,
                                          logits, mArr, vws);
  k_edge_gemm<1><<<200, 512, 0, stream>>>(nf, ei, shp, emb, elen, mrp,
                                          kw1, kb1, kw2, kb2, wdot, q,
                                          logits, mArr, vws);
  k_edge_pass2<<<(EE * HHD + 255) / 256, 256, 0, stream>>>(ei, logits, mArr, vws, den, agg);
  k_node_out<<<(NN + 7) / 8, 256, 0, stream>>>(nf, den, agg, wout, fw1, fw2, (float*)d_out);
}
